// Pocket_attention_75350906241277
// MI455X (gfx1250) — compile-verified, matrix operand order checked by reading, not run on hardware
//
#include <hip/hip_runtime.h>

// GAT layer for MI455X (gfx1250, wave32).
// Roofline: adjacency stream (128 MB) is the floor (~5.5us @ 23.3 TB/s).
// bf16 WMMA (16x16x32, f32 accum) keeps both GEMMs under that floor;
// fp32 WMMA (16x16x4, K=4) would be ~5x slower than memory on the P@h GEMM.
// K3 stages h_t B-fragments into LDS one chunk ahead with
// GLOBAL_LOAD_ASYNC_TO_LDS_B128 (ASYNCcnt) when the builtin is available.

typedef __bf16 bf16_t;
typedef int    v4i    __attribute__((ext_vector_type(4)));
typedef float  f32x4  __attribute__((ext_vector_type(4)));
typedef float  v8f    __attribute__((ext_vector_type(8)));
typedef bf16_t v16bf  __attribute__((ext_vector_type(16)));
typedef bf16_t v8bf   __attribute__((ext_vector_type(8)));
typedef bf16_t v4bf   __attribute__((ext_vector_type(4)));

constexpr int NB = 32;    // batch
constexpr int NN = 1024;  // nodes
constexpr int NF = 128;   // in features
constexpr int NH = 128;   // hidden

#if defined(__gfx1250__) && __has_builtin(__builtin_amdgcn_global_load_async_to_lds_b128)
#define USE_ASYNC_LDS 1
#else
#define USE_ASYNC_LDS 0
#endif

#if __has_builtin(__builtin_amdgcn_s_wait_asynccnt)
#define WAIT_ASYNC(n) __builtin_amdgcn_s_wait_asynccnt(n)
#else
#define WAIT_ASYNC(n) asm volatile("s_wait_asynccnt %0" ::"i"(n) : "memory")
#endif

__device__ __forceinline__ bf16_t f2bf(float f) { return (bf16_t)f; }
__device__ __forceinline__ float leaky02(float x) { return x > 0.f ? x : 0.2f * x; }

#if USE_ASYNC_LDS
// Async-copy 64 contiguous bytes (one h_t row chunk: 32 bf16) to LDS.
// Builtin signature (from compiler diagnostic): (int4*, int4*, imm offset, imm cpol).
__device__ __forceinline__ void async_h_chunk(const bf16_t* gsrc, bf16_t* ldst) {
    v4i* g = (v4i*)gsrc;
    v4i* l = (v4i*)ldst;
    __builtin_amdgcn_global_load_async_to_lds_b128(g, l, 0, 0);
    __builtin_amdgcn_global_load_async_to_lds_b128(g, l, 16, 0);
    __builtin_amdgcn_global_load_async_to_lds_b128(g, l, 32, 0);
    __builtin_amdgcn_global_load_async_to_lds_b128(g, l, 48, 0);
}
#endif

// ---------------------------------------------------------------------------
// Kernel 1: h = X @ W  (bf16 WMMA, f32 accum).
// Writes h transposed (h_t[b][hcol][node], bf16) + fp32 a_self/a_neigh.
// Block: 256 thr (8 waves). Tile: 64 nodes x 128 hidden.
// ---------------------------------------------------------------------------
__global__ __launch_bounds__(256)
void gat_h_kernel(const float* __restrict__ X, const float* __restrict__ W,
                  const float* __restrict__ Wself, const float* __restrict__ Wneigh,
                  bf16_t* __restrict__ h_t, float* __restrict__ a_self,
                  float* __restrict__ a_neigh)
{
    __shared__ __align__(16) bf16_t ldsWT[NH * NF];   // W^T: [n][k], 32 KB
    __shared__ __align__(16) float  ldsH[64 * NH];    // fp32 h tile, 32 KB

    const int b        = blockIdx.y;
    const int nodeBase = blockIdx.x * 64;
    const int t        = threadIdx.x;

    // Stage W^T as bf16 into LDS (W is [F][H] row-major).
    for (int idx = t; idx < NF * NH; idx += 256) {
        const int k = idx >> 7;          // F index
        const int n = idx & (NH - 1);    // H index
        ldsWT[n * NF + k] = f2bf(W[idx]);
    }
    __syncthreads();

    const int wave  = t >> 5;
    const int lane  = t & 31;
    const int strip = wave >> 1;          // 0..3
    const int nhalf = wave & 1;           // 0..1
    const int lr    = lane & 15;
    const int lhalf = lane >> 4;          // 0/1
    const int kb    = lhalf * 8;          // A/B frag: lane-half selects K+8

    const int    rowA = nodeBase + strip * 16 + lr;
    const float* Xrow = X + ((size_t)b * NN + rowA) * NF;

    v8f acc[4] = {v8f{}, v8f{}, v8f{}, v8f{}};

    for (int k0 = 0; k0 < NF; k0 += 32) {
        // A fragment: lane holds row lr, K = k0+kb..+7 and k0+kb+16..+23.
        v16bf afrag;
        {
            const f32x4 a0 = *(const f32x4*)(Xrow + k0 + kb);
            const f32x4 a1 = *(const f32x4*)(Xrow + k0 + kb + 4);
            const f32x4 a2 = *(const f32x4*)(Xrow + k0 + kb + 16);
            const f32x4 a3 = *(const f32x4*)(Xrow + k0 + kb + 20);
#pragma unroll
            for (int i = 0; i < 4; ++i) {
                afrag[i]      = f2bf(a0[i]);
                afrag[4 + i]  = f2bf(a1[i]);
                afrag[8 + i]  = f2bf(a2[i]);
                afrag[12 + i] = f2bf(a3[i]);
            }
        }
#pragma unroll
        for (int ti = 0; ti < 4; ++ti) {
            const int ncol = nhalf * 64 + ti * 16 + lr;
            const bf16_t* wp = &ldsWT[ncol * NF + k0 + kb];
            const v8bf b0 = *(const v8bf*)wp;
            const v8bf b1 = *(const v8bf*)(wp + 16);
            v16bf bfrag;
#pragma unroll
            for (int i = 0; i < 8; ++i) { bfrag[i] = b0[i]; bfrag[8 + i] = b1[i]; }
            acc[ti] = __builtin_amdgcn_wmma_f32_16x16x32_bf16(
                false, afrag, false, bfrag, (short)0, acc[ti], false, false);
        }
    }

    // Spill accumulators (C layout: row = v + lhalf*8, col = lr) to LDS fp32.
#pragma unroll
    for (int ti = 0; ti < 4; ++ti) {
        const int ncol = nhalf * 64 + ti * 16 + lr;
#pragma unroll
        for (int v = 0; v < 8; ++v) {
            const int row = strip * 16 + v + lhalf * 8;
            ldsH[row * NH + ncol] = acc[ti][v];
        }
    }
    __syncthreads();

    // fp32 attention projections before bf16 rounding.
    if (t < 64) {
        float s = 0.f, g = 0.f;
        for (int c = 0; c < NH; ++c) {
            const float hv = ldsH[t * NH + c];
            s += hv * Wself[c];
            g += hv * Wneigh[c];
        }
        a_self[b * NN + nodeBase + t]  = s;
        a_neigh[b * NN + nodeBase + t] = g;
    }

    // Store h transposed as bf16: h_t[b][hcol][node].
    for (int idx = t; idx < 64 * NH; idx += 256) {
        const int row  = idx & 63;
        const int hcol = idx >> 6;
        h_t[((size_t)b * NH + hcol) * NN + nodeBase + row] = f2bf(ldsH[row * NH + hcol]);
    }
}

// ---------------------------------------------------------------------------
// Kernel 2: per-batch max of a_neigh (softmax upper-bound trick).
// ---------------------------------------------------------------------------
__global__ __launch_bounds__(128)
void gat_amax_kernel(const float* __restrict__ a_neigh, float* __restrict__ amax)
{
    __shared__ float red[128];
    const int b = blockIdx.x;
    float m = -3.4e38f;
    for (int j = threadIdx.x; j < NN; j += 128)
        m = fmaxf(m, a_neigh[b * NN + j]);
    red[threadIdx.x] = m;
    __syncthreads();
    for (int s = 64; s > 0; s >>= 1) {
        if (threadIdx.x < s)
            red[threadIdx.x] = fmaxf(red[threadIdx.x], red[threadIdx.x + s]);
        __syncthreads();
    }
    if (threadIdx.x == 0) amax[b] = red[0];
}

// ---------------------------------------------------------------------------
// Kernel 3: softmax(attention) @ h, fused. Block: 128 thr (4 waves) owns a
// 16-row tile x all 128 hidden cols; streams adjacency in 16x32 chunks.
// Stable softmax via per-row bound M_i = leaky(a_self[i] + max_j a_neigh[j]).
// h_t B-fragments double-buffered in LDS via async copy (one chunk ahead).
// ---------------------------------------------------------------------------
__global__ __launch_bounds__(128)
void gat_attn_kernel(const float* __restrict__ adj, const float* __restrict__ mask,
                     const float* __restrict__ bias, const bf16_t* __restrict__ h_t,
                     const float* __restrict__ a_self, const float* __restrict__ a_neigh,
                     const float* __restrict__ amax, float* __restrict__ out)
{
    __shared__ __align__(16) bf16_t ldsP[2][16 * 32];     // P chunks, 2 KB
#if USE_ASYNC_LDS
    __shared__ __align__(16) bf16_t ldsHT[2][NH * 32];    // h_t chunks, 16 KB
#endif
    __shared__ float ldsRS[16 * 8];
    __shared__ float ldsRow[16];

    const int b       = blockIdx.y;
    const int rowBase = blockIdx.x * 16;
    const int t       = threadIdx.x;
    const int r       = t >> 3;    // 0..15: my logits row
    const int q       = t & 7;     // 0..7 : my 4-wide j slice
    const int node_i  = rowBase + r;

    const float as = a_self[b * NN + node_i];
    const float Mx = leaky02(as + amax[b]);          // row-wise logit upper bound
    const float* adjRow = adj + ((size_t)b * NN + node_i) * NN;
    const float* anRow  = a_neigh + b * NN;

    const int wave  = t >> 5;
    const int lane  = t & 31;
    const int lr    = lane & 15;
    const int lhalf = lane >> 4;
    const int kb    = lhalf * 8;

    const bf16_t* hb = h_t + (size_t)b * NH * NN;
    const int tiles[2] = {wave * 2, wave * 2 + 1};
#if !USE_ASYNC_LDS
    const bf16_t* hptr[2] = {hb + (size_t)(tiles[0] * 16 + lr) * NN,
                             hb + (size_t)(tiles[1] * 16 + lr) * NN};
#endif

    v8f acc[2] = {v8f{}, v8f{}};
    float psum = 0.f;

#if USE_ASYNC_LDS
    // Prologue: stage chunk jc=0. Thread t owns hcol=t (64 B per row chunk).
    async_h_chunk(hb + (size_t)t * NN, &ldsHT[0][t * 32]);
#endif

    for (int jc = 0; jc < NN; jc += 32) {
        const int buf = (jc >> 5) & 1;
        if (jc + 256 < NN) __builtin_prefetch(adjRow + jc + 256, 0, 1);

        const f32x4 av = *(const f32x4*)(adjRow + jc + q * 4);
        const f32x4 nv = *(const f32x4*)(anRow + jc + q * 4);
        v4bf pv;
#pragma unroll
        for (int i = 0; i < 4; ++i) {
            const float l = leaky02(as + nv[i]);
            const float p = (av[i] > 0.f) ? __expf(l - Mx) : 0.f;
            psum += p;
            pv[i] = f2bf(p);
        }
        *(v4bf*)&ldsP[buf][r * 32 + q * 4] = pv;

#if USE_ASYNC_LDS
        if (jc + 32 < NN) {   // uniform branch
            async_h_chunk(hb + (size_t)t * NN + jc + 32, &ldsHT[buf ^ 1][t * 32]);
            WAIT_ASYNC(4);    // previous chunk's 4 DMAs landed (in-order)
        } else {
            WAIT_ASYNC(0);
        }
#endif
        __syncthreads();

        // A fragment (same for all 4 waves): row lr, K = kb..+7 and kb+16..+23
        const v8bf a0 = *(const v8bf*)&ldsP[buf][lr * 32 + kb];
        const v8bf a1 = *(const v8bf*)&ldsP[buf][lr * 32 + kb + 16];
        v16bf afrag;
#pragma unroll
        for (int i = 0; i < 8; ++i) { afrag[i] = a0[i]; afrag[8 + i] = a1[i]; }

#pragma unroll
        for (int ti = 0; ti < 2; ++ti) {
#if USE_ASYNC_LDS
            const bf16_t* hp = &ldsHT[buf][(tiles[ti] * 16 + lr) * 32 + kb];
            const v8bf b0 = *(const v8bf*)hp;
            const v8bf b1 = *(const v8bf*)(hp + 16);
#else
            const v8bf b0 = *(const v8bf*)(hptr[ti] + jc + kb);
            const v8bf b1 = *(const v8bf*)(hptr[ti] + jc + kb + 16);
#endif
            v16bf bfrag;
#pragma unroll
            for (int i = 0; i < 8; ++i) { bfrag[i] = b0[i]; bfrag[8 + i] = b1[i]; }
            acc[ti] = __builtin_amdgcn_wmma_f32_16x16x32_bf16(
                false, afrag, false, bfrag, (short)0, acc[ti], false, false);
        }
        __syncthreads();   // guards ldsP/ldsHT buffer reuse next iteration
    }

    // Row-sum reduction (8 partials per row).
    ldsRS[r * 8 + q] = psum;
    __syncthreads();
    if (t < 16) {
        float s = 0.f;
#pragma unroll
        for (int i = 0; i < 8; ++i) s += ldsRS[t * 8 + i];
        ldsRow[t] = s;
    }
    __syncthreads();

    // Normalize + bias + mask, store.
#pragma unroll
    for (int ti = 0; ti < 2; ++ti) {
        const int hcol = tiles[ti] * 16 + lr;
        const float bv = bias[hcol];
#pragma unroll
        for (int v = 0; v < 8; ++v) {
            const int rr   = v + lhalf * 8;
            const int node = rowBase + rr;
            float val = acc[ti][v] / ldsRow[rr] + bv;
            val *= mask[b * NN + node];
            out[((size_t)b * NN + node) * NH + hcol] = val;
        }
    }
}

// ---------------------------------------------------------------------------
extern "C" void kernel_launch(void* const* d_in, const int* in_sizes, int n_in,
                              void* d_out, int out_size, void* d_ws, size_t ws_size,
                              hipStream_t stream)
{
    const float* X      = (const float*)d_in[0];  // [B,N,F]
    const float* adj    = (const float*)d_in[1];  // [B,N,N]
    const float* mask   = (const float*)d_in[2];  // [B,N]
    const float* W      = (const float*)d_in[3];  // [F,H]
    const float* bias   = (const float*)d_in[4];  // [H]
    const float* Wself  = (const float*)d_in[5];  // [H,1]
    const float* Wneigh = (const float*)d_in[6];  // [H,1]
    float* out = (float*)d_out;

    char* ws = (char*)d_ws;
    bf16_t* h_t    = (bf16_t*)ws;                               // 8 MB
    float*  a_self = (float*)(ws + (size_t)NB * NH * NN * sizeof(bf16_t));
    float*  a_neigh = a_self + NB * NN;                         // 128 KB each
    float*  amax    = a_neigh + NB * NN;                        // 128 B

    gat_h_kernel<<<dim3(NN / 64, NB), 256, 0, stream>>>(
        X, W, Wself, Wneigh, h_t, a_self, a_neigh);
    gat_amax_kernel<<<dim3(NB), 128, 0, stream>>>(a_neigh, amax);
    gat_attn_kernel<<<dim3(NN / 16, NB), 128, 0, stream>>>(
        adj, mask, bias, h_t, a_self, a_neigh, amax, out);
}
